// MultiHeadAttention_81630148428216
// MI455X (gfx1250) — matrix operand order repaired and checked by reading, not compiled, not run on hardware
//
#include <hip/hip_runtime.h>

#define B_  2
#define S_  2048
#define D_  1024
#define H_  16
#define DK_ 64

// GEMM block tile
#define BM 128
#define BN 128
#define BK 32

typedef __bf16 bf16;
typedef __attribute__((ext_vector_type(16))) __bf16 v16bf;
typedef __attribute__((ext_vector_type(8)))  __bf16 v8bf;
typedef __attribute__((ext_vector_type(8)))  float  v8f;
typedef __attribute__((ext_vector_type(4)))  unsigned int v4u;
typedef __attribute__((ext_vector_type(8)))  int v8i;
typedef __attribute__((ext_vector_type(4)))  int v4i;

#if __has_builtin(__builtin_amdgcn_tensor_load_to_lds)
#define HAVE_TDM 1
#endif

// ---------------------------------------------------------------------------
// f32 -> bf16 conversion (8 elements / thread)
// ---------------------------------------------------------------------------
__global__ __launch_bounds__(256) void cvt_f32_bf16(const float* __restrict__ in,
                                                    bf16* __restrict__ out, int n8) {
  int i = blockIdx.x * blockDim.x + threadIdx.x;
  if (i >= n8) return;
  const float* p = in + (size_t)i * 8;
  bf16* o = out + (size_t)i * 8;
#pragma unroll
  for (int j = 0; j < 8; ++j) o[j] = (bf16)p[j];
}

// ---------------------------------------------------------------------------
// Fragment helpers (contiguous-k operand order used by this code, wave32)
// A 16x32 (MxK): lanes 0-15 -> M=lane,   K = 0..15
//                lanes16-31 -> M=lane-16, K = 16..31
// B 32x16 (KxN): lanes 0-15 -> col N=lane,    K=0..15 (16 contiguous bf16)
//                lanes16-31 -> col N=lane-16, K=16..31
// ---------------------------------------------------------------------------
static __device__ __forceinline__ v16bf load_a16x32(const bf16* rowptr) {
  v16bf f;
  ((v8bf*)&f)[0] = *(const v8bf*)(rowptr);
  ((v8bf*)&f)[1] = *(const v8bf*)(rowptr + 8);
  return f;
}
static __device__ __forceinline__ v16bf load_b32x16(const bf16* p) {
  return *(const v16bf*)(p);
}

static __device__ __forceinline__ float rowmax16(float x) {
#pragma unroll
  for (int off = 8; off; off >>= 1) x = fmaxf(x, __shfl_xor(x, off, 16));
  return x;
}
static __device__ __forceinline__ float rowsum16(float x) {
#pragma unroll
  for (int off = 8; off; off >>= 1) x += __shfl_xor(x, off, 16);
  return x;
}

// ---------------------------------------------------------------------------
// CDNA5 async global->LDS staging (ASYNCcnt-tracked)
// ---------------------------------------------------------------------------
static __device__ __forceinline__ void async_load_b128(unsigned lds_off, const bf16* gptr) {
  asm volatile("global_load_async_to_lds_b128 %0, %1, off"
               :: "v"(lds_off), "v"(gptr) : "memory");
}
static __device__ __forceinline__ void wait_async() {
  asm volatile("s_wait_asynccnt 0x0" ::: "memory");
}

// Stage a 128-row x 32-col bf16 tile (8KB) = 512 x 16B chunks, 2 per thread
static __device__ __forceinline__ void stage_tile128(const bf16* src, int ld, int kk, bf16* dst) {
  unsigned base = (unsigned)(uintptr_t)dst;
#pragma unroll
  for (int j = 0; j < 2; ++j) {
    int chunk = (int)threadIdx.x + j * 256;
    int row = chunk >> 2;
    int qo  = (chunk & 3) * 8;   // element offset within the 32-wide row
    async_load_b128(base + (unsigned)(row * BK + qo) * 2,
                    src + (size_t)row * ld + kk + qo);
  }
}

#ifdef HAVE_TDM
// Stage W tile (BN=128 rows x BK=32 cols, 2B elems) via the Tensor Data Mover.
// 2D D# descriptor: tensor_dim0=K, tensor_dim0_stride=K, tile=32xBN, type=2.
static __device__ __forceinline__ void stage_W_tdm(const bf16* gsrc, int K, int N, bf16* dst) {
  unsigned lds = (unsigned)(uintptr_t)dst;
  unsigned long long ga = (unsigned long long)(uintptr_t)gsrc;
  v4u g0 = { 1u,                                   // count=1 (valid), user mode
             lds,                                  // lds_addr
             (unsigned)ga,                         // global_addr[31:0]
             (unsigned)((ga >> 32) & 0x1FFFFFFull) // global_addr[56:32]
               | (2u << 30) };                     // type=2 ("image")
  unsigned uK = (unsigned)K, uN = (unsigned)N;
  v8i g1 = { (int)(1u << 16),                              // data_size=1 (2B)
             (int)(uK << 16),                              // tensor_dim0 lo (bits 79:48)
             (int)(((uK >> 16) & 0xFFFFu) | (uN << 16)),   // tensor_dim0 hi | tensor_dim1 lo
             (int)(((uN >> 16) & 0xFFFFu) | ((unsigned)BK << 16)), // dim1 hi | tile_dim0
             (int)BN,                                      // tile_dim1 (tile_dim2=0)
             (int)uK,                                      // tensor_dim0_stride lo
             0, 0 };                                       // stride hi, dim1_stride = 0
  v4i z4 = {0, 0, 0, 0};
#if __clang_major__ >= 23
  v8i z8 = {0, 0, 0, 0, 0, 0, 0, 0};
  __builtin_amdgcn_tensor_load_to_lds(g0, g1, z4, z4, z8, 0);
#else
  __builtin_amdgcn_tensor_load_to_lds(g0, g1, z4, z4, 0);
#endif
}
#endif

// ---------------------------------------------------------------------------
// GEMM: Out(MxN) = A(MxK) @ W(NxK)^T + bias   (torch Linear)
// Block = 256 threads (8 waves) computes a BM x BN (128x128) tile.
// A tile staged via async-to-LDS, W tile staged via TDM, double buffered.
// Each wave: 1 A-fragment -> 8 WMMAs per K-step (two groups of 4, batched
// fragment loads so each group is one s_wait_dscnt + 4 back-to-back WMMAs).
// layout: 0 -> f32 (B,S,D); 1 -> bf16 (B,H,S,DK); 2 -> bf16 (B,H,DK,S)
// ---------------------------------------------------------------------------
__global__ __launch_bounds__(256)
void gemm_wmma(const bf16* __restrict__ A, const bf16* __restrict__ W,
               const float* __restrict__ bias, void* __restrict__ Out,
               int M, int N, int K, int layout) {
  __shared__ __align__(16) bf16 Asm[2][BM * BK];   // 2 x 8KB
  __shared__ __align__(16) bf16 Wsm[2][BN * BK];   // 2 x 8KB
  const int lane = threadIdx.x & 31;
  const int wave = threadIdx.x >> 5;
  const int Ntiles = N / BN;
  const int mt = blockIdx.x / Ntiles;
  const int nt = blockIdx.x % Ntiles;
  const int m0 = mt * BM, n0 = nt * BN;

  const bf16* Ablk = A + (size_t)m0 * K;
  const bf16* Wblk = W + (size_t)n0 * K;

  const int am  = lane & 15;
  const int ak  = (lane >> 4) << 4;   // 0 or 16
  const int bn  = lane & 15;
  const int bkh = (lane >> 4) << 4;   // 0 or 16

  // ---- prologue stage of buffer 0 ----
  stage_tile128(Ablk, K, 0, Asm[0]);
#ifdef HAVE_TDM
  if (wave == 0) stage_W_tdm(Wblk, K, N, Wsm[0]);
#else
  stage_tile128(Wblk, K, 0, Wsm[0]);
#endif
  wait_async();
#ifdef HAVE_TDM
  if (wave == 0) __builtin_amdgcn_s_wait_tensorcnt(0);
#endif
  __syncthreads();

  v8f acc[8];
#pragma unroll
  for (int t = 0; t < 8; ++t) acc[t] = (v8f){0.f,0.f,0.f,0.f,0.f,0.f,0.f,0.f};

  for (int kk = 0; kk < K; kk += BK) {
    const int cur = (kk >> 5) & 1;
    const int nxt = cur ^ 1;
    if (kk + BK < K) {          // issue next stage, wait later
      stage_tile128(Ablk, K, kk + BK, Asm[nxt]);
#ifdef HAVE_TDM
      if (wave == 0) stage_W_tdm(Wblk + kk + BK, K, N, Wsm[nxt]);
#else
      stage_tile128(Wblk, K, kk + BK, Wsm[nxt]);
#endif
    }
    const bf16* Al = Asm[cur];
    const bf16* Wl = Wsm[cur];
    v16bf af = load_a16x32(Al + (wave * 16 + am) * BK + ak);
#pragma unroll
    for (int g = 0; g < 2; ++g) {     // two groups of 4 N-tiles
      v16bf bfrag[4];
#pragma unroll
      for (int t = 0; t < 4; ++t)     // batch loads -> single dscnt wait
        bfrag[t] = load_b32x16(Wl + ((g * 4 + t) * 16 + bn) * BK + bkh);
#pragma unroll
      for (int t = 0; t < 4; ++t)     // 4 independent back-to-back WMMAs
        acc[g * 4 + t] = __builtin_amdgcn_wmma_f32_16x16x32_bf16(
            false, af, false, bfrag[t], (short)0, acc[g * 4 + t], false, false);
    }
    wait_async();
#ifdef HAVE_TDM
    if (wave == 0) __builtin_amdgcn_s_wait_tensorcnt(0);
#endif
    __syncthreads();
  }

  // ---- epilogue: C/D layout VGPR r -> row = mb + r, col = lane&15 ----
  const int n  = lane & 15;
  const int mb = (lane >> 4) << 3;
#pragma unroll
  for (int t = 0; t < 8; ++t) {
    const int col = n0 + t * 16 + n;
    const float bv = bias[col];
    if (layout == 0) {
      float* O = (float*)Out;
#pragma unroll
      for (int r = 0; r < 8; ++r)
        O[(size_t)(m0 + wave * 16 + mb + r) * N + col] = acc[t][r] + bv;
    } else {
      bf16* O = (bf16*)Out;
      const int h  = col >> 6;
      const int dk = col & 63;
#pragma unroll
      for (int r = 0; r < 8; ++r) {
        const int row = m0 + wave * 16 + mb + r;
        const int b   = row / S_;
        const int s   = row - b * S_;
        const float val = acc[t][r] + bv;
        if (layout == 1)  // (B,H,S,DK)
          O[(((size_t)b * H_ + h) * S_ + s) * DK_ + dk] = (bf16)val;
        else              // (B,H,DK,S)  (pre-transposed V)
          O[(((size_t)b * H_ + h) * DK_ + dk) * S_ + s] = (bf16)val;
      }
    }
  }
}

// ---------------------------------------------------------------------------
// Flash attention (causal), one wave per (b, h, 16-row query tile).
// Q,K: bf16 (B,H,S,DK) ; V: bf16 (B,H,DK,S) ; Ctx out: bf16 (B,S,D)
// ---------------------------------------------------------------------------
__global__ __launch_bounds__(256)
void attn_wmma(const bf16* __restrict__ Qp, const bf16* __restrict__ Kp,
               const bf16* __restrict__ Vt, bf16* __restrict__ Ctx) {
  const int lane = threadIdx.x & 31;
  const int wave = threadIdx.x >> 5;
  const int wid  = blockIdx.x * 8 + wave;
  const int QT   = S_ / 16;
  const int qt   = wid % QT;
  const int bh   = wid / QT;
  const int b    = bh >> 4;
  const int h    = bh & 15;
  const int q0   = qt << 4;

  __shared__ __align__(16) bf16 plds[8][16 * 32];   // P tile staging (C->A relayout)
  bf16* myp = plds[wave];

  const bf16* Qb = Qp + ((size_t)bh * S_ + q0) * DK_;
  const bf16* Kb = Kp + (size_t)bh * S_ * DK_;
  const bf16* Vb = Vt + (size_t)bh * DK_ * S_;

  const int am  = lane & 15;
  const int ak  = (lane >> 4) << 4;
  const int bn  = lane & 15;
  const int bkh = (lane >> 4) << 4;
  const int mb  = (lane >> 4) << 3;

  const bf16* qrow = Qb + am * DK_ + ak;
  v16bf qf0 = load_a16x32(qrow);
  v16bf qf1 = load_a16x32(qrow + 32);

  v8f o[4];
#pragma unroll
  for (int t = 0; t < 4; ++t) o[t] = (v8f){0.f,0.f,0.f,0.f,0.f,0.f,0.f,0.f};
  float mi[8], li[8];
#pragma unroll
  for (int r = 0; r < 8; ++r) { mi[r] = -1e30f; li[r] = 0.f; }

  const float scale = 0.125f;          // 1/sqrt(64)
  const int nchunks = (q0 >> 5) + 1;   // key chunks of 32

  for (int c = 0; c < nchunks; ++c) {
    const int kb = c << 5;
    if (c + 1 < nchunks) {             // prefetch next chunk of K and V
      __builtin_prefetch(Kb + (size_t)(kb + 32 + bn) * DK_, 0, 1);
      __builtin_prefetch(Vb + (size_t)bn * S_ + kb + 32, 0, 1);
    }
    // batch the 4 K-fragment loads, then 4 score WMMAs
    const bf16* k0 = Kb + (size_t)(kb + bn)      * DK_ + bkh;
    const bf16* k1 = Kb + (size_t)(kb + 16 + bn) * DK_ + bkh;
    v16bf kf00 = load_b32x16(k0);
    v16bf kf01 = load_b32x16(k0 + 32);
    v16bf kf10 = load_b32x16(k1);
    v16bf kf11 = load_b32x16(k1 + 32);
    v8f s0 = {0.f,0.f,0.f,0.f,0.f,0.f,0.f,0.f};
    v8f s1 = {0.f,0.f,0.f,0.f,0.f,0.f,0.f,0.f};
    s0 = __builtin_amdgcn_wmma_f32_16x16x32_bf16(false, qf0, false, kf00, (short)0, s0, false, false);
    s1 = __builtin_amdgcn_wmma_f32_16x16x32_bf16(false, qf0, false, kf10, (short)0, s1, false, false);
    s0 = __builtin_amdgcn_wmma_f32_16x16x32_bf16(false, qf1, false, kf01, (short)0, s0, false, false);
    s1 = __builtin_amdgcn_wmma_f32_16x16x32_bf16(false, qf1, false, kf11, (short)0, s1, false, false);

    const int kj0 = kb + bn;
    const int kj1 = kb + 16 + bn;
#pragma unroll
    for (int r = 0; r < 8; ++r) {
      const int qi = q0 + mb + r;
      float a0 = s0[r] * scale;  if (kj0 > qi) a0 = -1e9f;
      float a1 = s1[r] * scale;  if (kj1 > qi) a1 = -1e9f;
      float mcur = rowmax16(fmaxf(a0, a1));
      float mnew = fmaxf(mi[r], mcur);
      float alpha = __expf(mi[r] - mnew);
      float e0 = __expf(a0 - mnew);
      float e1 = __expf(a1 - mnew);
      float rs = rowsum16(e0 + e1);
      li[r] = li[r] * alpha + rs;
      mi[r] = mnew;
#pragma unroll
      for (int t = 0; t < 4; ++t) o[t][r] *= alpha;
      myp[(mb + r) * 32 + bn]      = (bf16)e0;
      myp[(mb + r) * 32 + 16 + bn] = (bf16)e1;
    }

    v16bf pf = load_a16x32(myp + am * 32 + ak);   // same-wave DS ops are in-order

    // batch the 4 V-fragment loads, then 4 PV WMMAs
    v16bf vf[4];
#pragma unroll
    for (int t = 0; t < 4; ++t)
      vf[t] = load_b32x16(Vb + (size_t)(t * 16 + bn) * S_ + kb + bkh);
#pragma unroll
    for (int t = 0; t < 4; ++t)
      o[t] = __builtin_amdgcn_wmma_f32_16x16x32_bf16(false, pf, false, vf[t],
                                                     (short)0, o[t], false, false);
  }

  float inv[8];
#pragma unroll
  for (int r = 0; r < 8; ++r) inv[r] = 1.0f / li[r];
  bf16* crow = Ctx + ((size_t)b * S_ + q0 + mb) * D_ + h * DK_ + bn;
#pragma unroll
  for (int t = 0; t < 4; ++t)
#pragma unroll
    for (int r = 0; r < 8; ++r)
      crow[(size_t)r * D_ + t * 16] = (bf16)(o[t][r] * inv[r]);
}

// ---------------------------------------------------------------------------
// Host launcher
// ---------------------------------------------------------------------------
extern "C" void kernel_launch(void* const* d_in, const int* in_sizes, int n_in,
                              void* d_out, int out_size, void* d_ws, size_t ws_size,
                              hipStream_t stream) {
  const float* q    = (const float*)d_in[0];
  const float* k    = (const float*)d_in[1];
  const float* v    = (const float*)d_in[2];
  // d_in[3] = causal mask (tril) — applied analytically in attn_wmma
  const float* wq_w = (const float*)d_in[4];
  const float* wq_b = (const float*)d_in[5];
  const float* wk_w = (const float*)d_in[6];
  const float* wk_b = (const float*)d_in[7];
  const float* wv_w = (const float*)d_in[8];
  const float* wv_b = (const float*)d_in[9];
  const float* wo_w = (const float*)d_in[10];
  const float* wo_b = (const float*)d_in[11];
  float* out = (float*)d_out;

  const size_t NTOK = (size_t)B_ * S_;     // 4096
  const size_t ACT  = NTOK * D_;           // 4,194,304 elems
  const size_t WEL  = (size_t)D_ * D_;     // 1,048,576 elems

  char* ws = (char*)d_ws;
  bf16* xq  = (bf16*)ws; ws += ACT * 2;
  bf16* xk  = (bf16*)ws; ws += ACT * 2;
  bf16* xv  = (bf16*)ws; ws += ACT * 2;
  bf16* bwq = (bf16*)ws; ws += WEL * 2;
  bf16* bwk = (bf16*)ws; ws += WEL * 2;
  bf16* bwv = (bf16*)ws; ws += WEL * 2;
  bf16* bwo = (bf16*)ws; ws += WEL * 2;
  bf16* Qp  = (bf16*)ws; ws += ACT * 2;    // (B,H,S,DK)
  bf16* Kp  = (bf16*)ws; ws += ACT * 2;    // (B,H,S,DK)
  bf16* Vtp = (bf16*)ws; ws += ACT * 2;    // (B,H,DK,S)
  bf16* Ctx = (bf16*)ws; ws += ACT * 2;    // (B,S,D)

  const int cvtA = (int)(ACT / 8 / 256);
  const int cvtW = (int)(WEL / 8 / 256);
  cvt_f32_bf16<<<cvtA, 256, 0, stream>>>(q, xq, (int)(ACT / 8));
  cvt_f32_bf16<<<cvtA, 256, 0, stream>>>(k, xk, (int)(ACT / 8));
  cvt_f32_bf16<<<cvtA, 256, 0, stream>>>(v, xv, (int)(ACT / 8));
  cvt_f32_bf16<<<cvtW, 256, 0, stream>>>(wq_w, bwq, (int)(WEL / 8));
  cvt_f32_bf16<<<cvtW, 256, 0, stream>>>(wk_w, bwk, (int)(WEL / 8));
  cvt_f32_bf16<<<cvtW, 256, 0, stream>>>(wv_w, bwv, (int)(WEL / 8));
  cvt_f32_bf16<<<cvtW, 256, 0, stream>>>(wo_w, bwo, (int)(WEL / 8));

  const int gblocks = (int)((NTOK / BM) * (D_ / BN));  // 32*8 = 256
  gemm_wmma<<<gblocks, 256, 0, stream>>>(xq, bwq, wq_b, Qp,  (int)NTOK, D_, D_, 1);
  gemm_wmma<<<gblocks, 256, 0, stream>>>(xk, bwk, wk_b, Kp,  (int)NTOK, D_, D_, 1);
  gemm_wmma<<<gblocks, 256, 0, stream>>>(xv, bwv, wv_b, Vtp, (int)NTOK, D_, D_, 2);

  const int attnBlocks = (B_ * H_ * (S_ / 16)) / 8;    // 512
  attn_wmma<<<attnBlocks, 256, 0, stream>>>(Qp, Kp, Vtp, Ctx);

  gemm_wmma<<<gblocks, 256, 0, stream>>>(Ctx, bwo, wo_b, out, (int)NTOK, D_, D_, 0);
}
